// Decoder_60610578482078
// MI455X (gfx1250) — matrix operand order repaired and checked by reading, not compiled, not run on hardware
//
#include <hip/hip_runtime.h>
#include <hip/hip_bf16.h>

// ---------------------------------------------------------------------------
// CDNA5 (gfx1250, wave32) transformer decoder.
// f32 inputs are converted ONCE to f16; every GEMM is f16 x f16 -> f32 acc via
// v_wmma_f32_16x16x32_f16 with double-buffered LDS staged by
// global_load_async_to_lds_b128 (ASYNCcnt) on the fast path.
// Fragment layouts per CDNA5 ISA:
//   A 16x32 f16 : lanes 0-15 -> M=lane, K-halves {0..7,16..23};
//                 lanes 16-31 -> M=lane-16, K-halves {8..15,24..31}
//   B 32x16 f16 : lane L -> N=L%16, same K-halves as A
//   C 16x16 f32 : VGPR v -> (M = v + 8*(lane>=16), N = lane%16)
// ---------------------------------------------------------------------------

typedef _Float16 v16h __attribute__((ext_vector_type(16)));
typedef _Float16 v8h  __attribute__((ext_vector_type(8)));
typedef float    v8f  __attribute__((ext_vector_type(8)));

#define TM 64
#define TN 64
#define TK 32

// ---- gfx1250 async copy: global -> LDS, 16 bytes per lane ------------------
__device__ __forceinline__ unsigned ldsoff(const void* p)
{
    return (unsigned)(unsigned long long)p;     // low 32 bits = LDS address
}
__device__ __forceinline__ void async_b128(const void* lds, const void* g)
{
    asm volatile("global_load_async_to_lds_b128 %0, %1, off"
                 :: "v"(ldsoff(lds)), "v"((unsigned long long)g) : "memory");
}
__device__ __forceinline__ void wait_async0()
{
    asm volatile("s_wait_asynccnt 0" ::: "memory");
}

// ---- staging helpers (f16 sources) -----------------------------------------

// A tile: TM x TK (64x32) halves = 256 x 16B chunks, one async b128 per thread
__device__ __forceinline__
void stageA(const _Float16* __restrict__ A, _Float16 (*dst)[TK],
            int tileM, int k0, int M, int K, int lda, int tid, bool full)
{
    if (full) {
        int r  = tid >> 2;
        int kc = (tid & 3) * 8;
        async_b128(&dst[r][kc], &A[(long long)(tileM + r) * lda + k0 + kc]);
    } else {
        #pragma unroll
        for (int t = 0; t < 8; ++t) {
            int idx = tid + t * 256;
            int r = idx >> 5, c = idx & 31;
            int gm = tileM + r, gk = k0 + c;
            dst[r][c] = (gm < M && gk < K) ? A[(long long)gm * lda + gk]
                                           : (_Float16)0.f;
        }
    }
}

// B tile: TK x TN (32x64) halves, stored [k][n]
__device__ __forceinline__
void stageB(const _Float16* __restrict__ B, _Float16 (*dst)[TN],
            int tileN, int k0, int N, int K, int ldb, int tid,
            bool transB, bool full)
{
    if (!transB) {
        if (full) {
            int r = tid >> 3;
            int c = (tid & 7) * 8;
            async_b128(&dst[r][c], &B[(long long)(k0 + r) * ldb + tileN + c]);
        } else {
            #pragma unroll
            for (int t = 0; t < 8; ++t) {
                int idx = tid + t * 256;
                int r = idx >> 6, c = idx & 63;
                int gk = k0 + r, gn = tileN + c;
                dst[r][c] = (gk < K && gn < N) ? B[(long long)gk * ldb + gn]
                                               : (_Float16)0.f;
            }
        }
    } else {
        // B stored [N,K]; vector-load 8 halves along K, scatter into [k][n]
        if (full) {
            int n  = tid >> 2;
            int kc = (tid & 3) * 8;
            v8h v = *(const v8h*)&B[(long long)(tileN + n) * ldb + k0 + kc];
            #pragma unroll
            for (int i = 0; i < 8; ++i) dst[kc + i][n] = v[i];
        } else {
            #pragma unroll
            for (int t = 0; t < 8; ++t) {
                int idx = tid + t * 256;
                int n = idx >> 5, c = idx & 31;
                int gk = k0 + c, gn = tileN + n;
                dst[c][n] = (gk < K && gn < N) ? B[(long long)gn * ldb + gk]
                                               : (_Float16)0.f;
            }
        }
    }
}

// ---- GEMM ------------------------------------------------------------------
// flags: 1=transB, 2=relu, 4=bias, 8=write f32 C, 16=write f16 C
__global__ __launch_bounds__(256)
void gemm_f16_wmma(const _Float16* __restrict__ A, const _Float16* __restrict__ B,
                   const float* __restrict__ bias,
                   float* __restrict__ Cf, _Float16* __restrict__ Ch,
                   int M, int N, int K, int lda, int ldb, int ldc,
                   long long offA_b, long long offA_h,
                   long long offB_b, long long offB_h,
                   long long offC_b, long long offC_h,
                   int Hdiv, float alpha, int flags)
{
    __shared__ __align__(16) _Float16 sA[2][TM][TK];
    __shared__ __align__(16) _Float16 sB[2][TK][TN];

    const int tid  = threadIdx.x;
    const int wave = tid >> 5;
    const int lane = tid & 31;

    const int z  = blockIdx.z;
    const int bb = z / Hdiv;
    const int hh = z % Hdiv;
    A  += (long long)bb * offA_b + (long long)hh * offA_h;
    B  += (long long)bb * offB_b + (long long)hh * offB_h;
    const long long coff = (long long)bb * offC_b + (long long)hh * offC_h;

    const int tileM = blockIdx.y * TM;
    const int tileN = blockIdx.x * TN;
    const int wm = (wave >> 1) * 16;          // 4 wave rows in M
    const int wn = (wave & 1) * 32;           // 2 wave cols, 32 wide each
    const bool transB = (flags & 1) != 0;
    const bool fullA  = (tileM + TM <= M);    // K is a multiple of TK in all uses
    const bool fullB  = (tileN + TN <= N);

    v8f acc0 = {}, acc1 = {};

    stageA(A, sA[0], tileM, 0, M, K, lda, tid, fullA);
    stageB(B, sB[0], tileN, 0, N, K, ldb, tid, transB, fullB);
    wait_async0();
    __syncthreads();

    int cur = 0;
    for (int k0 = 0; k0 < K; k0 += TK) {
        int nxt = k0 + TK;
        if (nxt < K) {                        // async-prefetch next tiles
            stageA(A, sA[cur ^ 1], tileM, nxt, M, K, lda, tid, fullA);
            stageB(B, sB[cur ^ 1], tileN, nxt, N, K, ldb, tid, transB, fullB);
        }

        // A fragment: two ds_load_b128
        const int mrow = wm + (lane & 15);
        const int ka   = (lane >> 4) ? 8 : 0;
        v8h alo = *(const v8h*)&sA[cur][mrow][ka];
        v8h ahi = *(const v8h*)&sA[cur][mrow][ka + 16];
        v16h afrag;
        #pragma unroll
        for (int i = 0; i < 8; ++i) { afrag[i] = alo[i]; afrag[i + 8] = ahi[i]; }

        // B fragments for two 16-wide N sub-tiles (reuse A)
        const int ncol = wn + (lane & 15);
        v16h bf0, bf1;
        #pragma unroll
        for (int i = 0; i < 8; ++i) {
            bf0[i] = sB[cur][ka + i][ncol]; bf0[i + 8] = sB[cur][ka + 16 + i][ncol];
            bf1[i] = sB[cur][ka + i][ncol + 16]; bf1[i + 8] = sB[cur][ka + 16 + i][ncol + 16];
        }

        acc0 = __builtin_amdgcn_wmma_f32_16x16x32_f16(
                   false, afrag, false, bf0, (short)0, acc0, false, false);
        acc1 = __builtin_amdgcn_wmma_f32_16x16x32_f16(
                   false, afrag, false, bf1, (short)0, acc1, false, false);

        wait_async0();
        __syncthreads();
        cur ^= 1;
    }

    // ---- epilogue (documented 16x16 f32 C layout) ----
    const int n0 = tileN + wn + (lane & 15);
    const int n1 = n0 + 16;
    const int mbase = tileM + wm + ((lane >> 4) << 3);
    const bool dobias = (flags & 4) != 0;
    const bool dorelu = (flags & 2) != 0;
    float bv0 = (dobias && n0 < N) ? bias[n0] : 0.f;
    float bv1 = (dobias && n1 < N) ? bias[n1] : 0.f;
    #pragma unroll
    for (int v = 0; v < 8; ++v) {
        int m = mbase + v;
        if (m < M) {
            long long base = coff + (long long)m * ldc;
            if (n0 < N) {
                float val = alpha * acc0[v] + bv0;
                if (dorelu) val = val > 0.f ? val : 0.f;
                if (flags & 8)  Cf[base + n0] = val;
                if (flags & 16) Ch[base + n0] = (_Float16)val;
            }
            if (n1 < N) {
                float val = alpha * acc1[v] + bv1;
                if (dorelu) val = val > 0.f ? val : 0.f;
                if (flags & 8)  Cf[base + n1] = val;
                if (flags & 16) Ch[base + n1] = (_Float16)val;
            }
        }
    }
}

// ---- elementwise / reduction kernels ---------------------------------------

__global__ void f32_to_f16(const float* __restrict__ src,
                           _Float16* __restrict__ dst, long long n)
{
    long long i = (long long)blockIdx.x * blockDim.x + threadIdx.x;
    if (i < n) dst[i] = (_Float16)src[i];
}

// x = emb[ids] * scale (f32 + f16 copies)
__global__ void embed_kernel(const int* __restrict__ ids,
                             const float* __restrict__ emb,
                             float* __restrict__ x, _Float16* __restrict__ xh,
                             int ntok, int D, float scale)
{
    int i = blockIdx.x * blockDim.x + threadIdx.x;
    if (i < ntok * D) {
        int t = i / D, d = i % D;
        float v = emb[(long long)ids[t] * D + d] * scale;
        x[i] = v;
        xh[i] = (_Float16)v;
    }
}

// per-channel mean / biased var of (X + S) over rows
__global__ __launch_bounds__(256)
void bn_stats(const float* __restrict__ X, const float* __restrict__ S,
              float* __restrict__ mu, float* __restrict__ var, int rows, int D)
{
    __shared__ float s1[256], s2[256];
    int d = blockIdx.x;
    float sum = 0.f, ss = 0.f;
    for (int r = threadIdx.x; r < rows; r += 256) {
        float v = X[(long long)r * D + d] + S[(long long)r * D + d];
        sum += v; ss += v * v;
    }
    s1[threadIdx.x] = sum; s2[threadIdx.x] = ss; __syncthreads();
    for (int o = 128; o > 0; o >>= 1) {
        if (threadIdx.x < o) {
            s1[threadIdx.x] += s1[threadIdx.x + o];
            s2[threadIdx.x] += s2[threadIdx.x + o];
        }
        __syncthreads();
    }
    if (threadIdx.x == 0) {
        float m = s1[0] / (float)rows;
        mu[d] = m;
        var[d] = s2[0] / (float)rows - m * m;
    }
}

// X = ((X + S) - mu) * rsqrt(var + eps) * g + b  (in place) ; also f16 copy
__global__ void bn_apply(float* __restrict__ X, const float* __restrict__ S,
                         const float* __restrict__ mu, const float* __restrict__ var,
                         const float* __restrict__ g, const float* __restrict__ b,
                         _Float16* __restrict__ XH, int rows, int D)
{
    int i = blockIdx.x * blockDim.x + threadIdx.x;
    if (i < rows * D) {
        int d = i % D;
        float v = X[i] + S[i];
        v = (v - mu[d]) * rsqrtf(var[d] + 1e-5f) * g[d] + b[d];
        X[i] = v;
        XH[i] = (_Float16)v;
    }
}

// row softmax on f32 X; causal limits live entries to (row%S)+1.
// mode 0: write back f32 in place.  mode 1: write f16 to OH (X untouched).
__global__ __launch_bounds__(256)
void softmax_rows(float* __restrict__ X, _Float16* __restrict__ OH,
                  int N, int S, int causal, int mode)
{
    __shared__ float red[256];
    long long row = blockIdx.x;
    float* p = X + row * (long long)N;
    int limit = causal ? (int)(row % S) + 1 : N;

    float mx = -3.4e38f;
    for (int j = threadIdx.x; j < limit; j += 256) mx = fmaxf(mx, p[j]);
    red[threadIdx.x] = mx; __syncthreads();
    for (int o = 128; o > 0; o >>= 1) {
        if (threadIdx.x < o) red[threadIdx.x] = fmaxf(red[threadIdx.x], red[threadIdx.x + o]);
        __syncthreads();
    }
    mx = red[0]; __syncthreads();

    float sum = 0.f;
    for (int j = threadIdx.x; j < limit; j += 256) sum += __expf(p[j] - mx);
    red[threadIdx.x] = sum; __syncthreads();
    for (int o = 128; o > 0; o >>= 1) {
        if (threadIdx.x < o) red[threadIdx.x] += red[threadIdx.x + o];
        __syncthreads();
    }
    float inv = 1.f / red[0];

    if (mode == 0) {
        for (int j = threadIdx.x; j < N; j += 256)
            p[j] = (j < limit) ? __expf(p[j] - mx) * inv : 0.f;
    } else {
        _Float16* q = OH + row * (long long)N;
        for (int j = threadIdx.x; j < N; j += 256)
            q[j] = (j < limit) ? (_Float16)(__expf(p[j] - mx) * inv) : (_Float16)0.f;
    }
}

// ---------------------------------------------------------------------------

static inline void gemm(hipStream_t st, const _Float16* A, const _Float16* B,
                        const float* bias, float* Cf, _Float16* Ch,
                        int M, int N, int K, int lda, int ldb, int ldc,
                        long long oAb, long long oAh, long long oBb, long long oBh,
                        long long oCb, long long oCh,
                        int nbatch, int Hdiv, float alpha, int flags)
{
    dim3 g((N + TN - 1) / TN, (M + TM - 1) / TM, nbatch);
    gemm_f16_wmma<<<g, dim3(256), 0, st>>>(A, B, bias, Cf, Ch, M, N, K, lda, ldb, ldc,
                                           oAb, oAh, oBb, oBh, oCb, oCh,
                                           Hdiv, alpha, flags);
}

static inline void cvt(hipStream_t st, const float* s, _Float16* d, long long n)
{
    f32_to_f16<<<(unsigned)((n + 255) / 256), 256, 0, st>>>(s, d, n);
}

extern "C" void kernel_launch(void* const* d_in, const int* in_sizes, int n_in,
                              void* d_out, int out_size, void* d_ws, size_t ws_size,
                              hipStream_t stream)
{
    const int Bn = 4, S = 512, SE = 512, D = 512, H = 8, DK = 64;
    const int NX = 6, DFF = 2048, V = 37000;
    const int NT = Bn * S;                               // 2048 tokens

    const int*   ids   = (const int*)  d_in[0];
    const float* enc   = (const float*)d_in[1];
    const float* emb   = (const float*)d_in[2];
    const float* Wq1   = (const float*)d_in[3];
    const float* Wk1   = (const float*)d_in[4];
    const float* Wv1   = (const float*)d_in[5];
    const float* Wo1   = (const float*)d_in[6];
    const float* Wq2   = (const float*)d_in[7];
    const float* Wk2   = (const float*)d_in[8];
    const float* Wv2   = (const float*)d_in[9];
    const float* Wo2   = (const float*)d_in[10];
    const float* gamma = (const float*)d_in[11];
    const float* beta  = (const float*)d_in[12];
    const float* W1    = (const float*)d_in[13];
    const float* b1    = (const float*)d_in[14];
    const float* W2    = (const float*)d_in[15];
    const float* b2    = (const float*)d_in[16];
    const float* Wout  = (const float*)d_in[17];
    const float* bout  = (const float*)d_in[18];
    float*       out   = (float*)d_out;

    const long long SZ_ACT = (long long)NT * D;          // 1,048,576
    const long long SZ_PRJ = (long long)NX * H * D * DK; // 1,572,864 per tensor
    const long long SZ_FF  = (long long)NX * D * DFF;    // 6,291,456

    // ---- d_ws: f32 state + f16 activations + f16 Wout (~68 MB) ----
    float* ws   = (float*)d_ws;
    float* xbuf = ws;                                    // f32 [2048,512]
    float* sub  = xbuf + SZ_ACT;                         // f32 sublayer out
    float* mu   = sub + SZ_ACT;
    float* var  = mu + D;
    _Float16* h    = (_Float16*)(var + D + 62);          // offset 8 mod 16 B from d_ws
    _Float16* wouth = h;          h += (long long)D * V; // 18,944,000
    _Float16* ench  = h;          h += SZ_ACT;
    _Float16* xh    = h;          h += SZ_ACT;
    _Float16* qh    = h;          h += SZ_ACT;
    _Float16* kh    = h;          h += SZ_ACT;
    _Float16* vh    = h;          h += SZ_ACT;
    _Float16* attch = h;          h += SZ_ACT;
    _Float16* ffh   = h;          h += (long long)NT * DFF;

    // ---- d_out scratch (dead before final GEMM overwrites it) ----
    float* scores = out;                                 // f32 [B,H,S,S]
    _Float16* oh  = (_Float16*)(out + (long long)Bn * H * S * S);
    _Float16* probsh = oh;        oh += (long long)Bn * H * S * S;
    _Float16* wq1h = oh;          oh += SZ_PRJ;
    _Float16* wk1h = oh;          oh += SZ_PRJ;
    _Float16* wv1h = oh;          oh += SZ_PRJ;
    _Float16* wo1h = oh;          oh += SZ_PRJ;
    _Float16* wq2h = oh;          oh += SZ_PRJ;
    _Float16* wk2h = oh;          oh += SZ_PRJ;
    _Float16* wv2h = oh;          oh += SZ_PRJ;
    _Float16* wo2h = oh;          oh += SZ_PRJ;
    _Float16* w1h  = oh;          oh += SZ_FF;
    _Float16* w2h  = oh;          oh += SZ_FF;

    // ---- one-time f32 -> f16 conversions ----
    cvt(stream, enc,  ench, SZ_ACT);
    cvt(stream, Wq1,  wq1h, SZ_PRJ);  cvt(stream, Wk1, wk1h, SZ_PRJ);
    cvt(stream, Wv1,  wv1h, SZ_PRJ);  cvt(stream, Wo1, wo1h, SZ_PRJ);
    cvt(stream, Wq2,  wq2h, SZ_PRJ);  cvt(stream, Wk2, wk2h, SZ_PRJ);
    cvt(stream, Wv2,  wv2h, SZ_PRJ);  cvt(stream, Wo2, wo2h, SZ_PRJ);
    cvt(stream, W1,   w1h,  SZ_FF);   cvt(stream, W2,  w2h,  SZ_FF);
    cvt(stream, Wout, wouth, (long long)D * V);

    // ---- embedding (x = emb[ids] * sqrt(d_k)) ----
    embed_kernel<<<(NT * D) / 256, 256, 0, stream>>>(ids, emb, xbuf, xh, NT, D, 8.0f);

    const long long strQ_b = (long long)H * S * DK, strQ_h = (long long)S * DK;
    const long long strS_b = (long long)H * S * S,  strS_h = (long long)S * S;

    for (int i = 0; i < NX; ++i) {
        const long long oP = (long long)i * H * D * DK;
        // ================= self-attention (causal) =================
        gemm(stream, xh, wq1h + oP, nullptr, nullptr, qh, S, DK, D, D, DK, DK,
             (long long)S * D, 0, 0, (long long)D * DK, strQ_b, strQ_h, Bn * H, H, 1.f, 16);
        gemm(stream, xh, wk1h + oP, nullptr, nullptr, kh, S, DK, D, D, DK, DK,
             (long long)S * D, 0, 0, (long long)D * DK, strQ_b, strQ_h, Bn * H, H, 1.f, 16);
        gemm(stream, xh, wv1h + oP, nullptr, nullptr, vh, S, DK, D, D, DK, DK,
             (long long)S * D, 0, 0, (long long)D * DK, strQ_b, strQ_h, Bn * H, H, 1.f, 16);
        gemm(stream, qh, kh, nullptr, scores, nullptr, S, S, DK, DK, DK, S,
             strQ_b, strQ_h, strQ_b, strQ_h, strS_b, strS_h, Bn * H, H, 0.125f, 1 | 8);
        softmax_rows<<<Bn * H * S, 256, 0, stream>>>(scores, probsh, S, S, 1, 1);
        gemm(stream, probsh, vh, nullptr, nullptr, attch, S, DK, S, S, DK, D,
             strS_b, strS_h, strQ_b, strQ_h, (long long)S * D, (long long)DK, Bn * H, H, 1.f, 16);
        gemm(stream, attch, wo1h + (long long)i * D * D, nullptr, sub, nullptr,
             NT, D, D, D, D, D, 0, 0, 0, 0, 0, 0, 1, 1, 1.f, 8);
        bn_stats<<<D, 256, 0, stream>>>(xbuf, sub, mu, var, NT, D);
        bn_apply<<<(NT * D) / 256, 256, 0, stream>>>(xbuf, sub, mu, var,
                 gamma + (long long)(i * 3 + 0) * D, beta + (long long)(i * 3 + 0) * D, xh, NT, D);

        // ================= cross-attention =================
        gemm(stream, xh,   wq2h + oP, nullptr, nullptr, qh, S,  DK, D, D, DK, DK,
             (long long)S * D, 0, 0, (long long)D * DK, strQ_b, strQ_h, Bn * H, H, 1.f, 16);
        gemm(stream, ench, wk2h + oP, nullptr, nullptr, kh, SE, DK, D, D, DK, DK,
             (long long)SE * D, 0, 0, (long long)D * DK, strQ_b, strQ_h, Bn * H, H, 1.f, 16);
        gemm(stream, ench, wv2h + oP, nullptr, nullptr, vh, SE, DK, D, D, DK, DK,
             (long long)SE * D, 0, 0, (long long)D * DK, strQ_b, strQ_h, Bn * H, H, 1.f, 16);
        gemm(stream, qh, kh, nullptr, scores, nullptr, S, SE, DK, DK, DK, SE,
             strQ_b, strQ_h, strQ_b, strQ_h, strS_b, strS_h, Bn * H, H, 0.125f, 1 | 8);
        softmax_rows<<<Bn * H * S, 256, 0, stream>>>(scores, probsh, SE, S, 0, 1);
        gemm(stream, probsh, vh, nullptr, nullptr, attch, S, DK, SE, SE, DK, D,
             strS_b, strS_h, strQ_b, strQ_h, (long long)S * D, (long long)DK, Bn * H, H, 1.f, 16);
        gemm(stream, attch, wo2h + (long long)i * D * D, nullptr, sub, nullptr,
             NT, D, D, D, D, D, 0, 0, 0, 0, 0, 0, 1, 1, 1.f, 8);
        bn_stats<<<D, 256, 0, stream>>>(xbuf, sub, mu, var, NT, D);
        bn_apply<<<(NT * D) / 256, 256, 0, stream>>>(xbuf, sub, mu, var,
                 gamma + (long long)(i * 3 + 1) * D, beta + (long long)(i * 3 + 1) * D, xh, NT, D);

        // ================= feed-forward =================
        gemm(stream, xh, w1h + (long long)i * D * DFF, b1 + (long long)i * DFF, nullptr, ffh,
             NT, DFF, D, D, DFF, DFF, 0, 0, 0, 0, 0, 0, 1, 1, 1.f, 16 | 4 | 2);
        gemm(stream, ffh, w2h + (long long)i * DFF * D, b2 + (long long)i * D, sub, nullptr,
             NT, D, DFF, DFF, D, D, 0, 0, 0, 0, 0, 0, 1, 1, 1.f, 8 | 4);
        bn_stats<<<D, 256, 0, stream>>>(xbuf, sub, mu, var, NT, D);
        bn_apply<<<(NT * D) / 256, 256, 0, stream>>>(xbuf, sub, mu, var,
                 gamma + (long long)(i * 3 + 2) * D, beta + (long long)(i * 3 + 2) * D, xh, NT, D);
    }

    // ---- final projection + softmax over vocab (overwrites all scratch) ----
    gemm(stream, xh, wouth, bout, out, nullptr, NT, V, D, D, V, V,
         0, 0, 0, 0, 0, 0, 1, 1, 1.f, 8 | 4);
    softmax_rows<<<NT, 256, 0, stream>>>(out, nullptr, V, V, 0, 0);

    (void)in_sizes; (void)n_in; (void)out_size; (void)ws_size;
}
